// SelectivePooling_41592463294659
// MI455X (gfx1250) — matrix operand order repaired and checked by reading, not compiled, not run on hardware
//
#include <hip/hip_runtime.h>

// ---------------------------------------------------------------------------
// SelectivePooling on MI455X (gfx1250, wave32).
//   B=128, H=W=28 -> N=784 pixels, C=512 channels, K=64 anchors.
//   Pipeline:
//     1) sp_norm_anchors : kernel(64x512 f32) -> unit-norm bf16 table (64KB)
//     2) sp_gemm_logA    : bf16 WMMA cosine GEMM + fused per-pixel
//                          logA[b,n] = logsumexp_k( (2*cos-2)/eps )
//     3) sp_sinkhorn     : 10-step scalar v-recursion per batch ->
//                          w[b,n] = sigmoid(v/eps+logA)/(mu*N)
//     4) sp_wsum_partial : chunked weighted sum over N  (deterministic)
//     5) sp_reduce_out   : sum the 4 chunk partials -> emb (128x512)
// ---------------------------------------------------------------------------

#define NPOS   784
#define CCH    512
#define NANCH  64
#define EPS_INV 10.0f   // 1/EPS_REG

typedef __attribute__((ext_vector_type(16))) __bf16 v16bf;
typedef __attribute__((ext_vector_type(16))) float  v16f;
typedef __attribute__((ext_vector_type(8)))  float  v8f;

// --------------------------- 1) anchor normalize ---------------------------
__global__ void sp_norm_anchors(const float* __restrict__ kern,
                                unsigned short* __restrict__ anbf) {
  const int a   = blockIdx.x;       // anchor row
  const int tid = threadIdx.x;      // 256 threads, 2 channels each
  float v0 = kern[a * CCH + tid];
  float v1 = kern[a * CCH + tid + 256];
  float ss = fmaf(v0, v0, v1 * v1);
  #pragma unroll
  for (int off = 16; off > 0; off >>= 1) ss += __shfl_xor(ss, off, 32);
  __shared__ float red[8];
  if ((tid & 31) == 0) red[tid >> 5] = ss;
  __syncthreads();
  float tot = 0.f;
  #pragma unroll
  for (int w = 0; w < 8; ++w) tot += red[w];
  const float inv = 1.f / (sqrtf(tot) + 1e-12f);
  anbf[a * CCH + tid] =
      __builtin_bit_cast(unsigned short, (__bf16)(v0 * inv));
  anbf[a * CCH + tid + 256] =
      __builtin_bit_cast(unsigned short, (__bf16)(v1 * inv));
}

// ------------------- 2) WMMA GEMM + fused logsumexp ------------------------
// Grid: (13, 128).  Block: 128 threads = 4 waves.  Wave w owns rows
// [blk*64 + w*16, +16) x all 64 anchors, K-loop over C in steps of 32.
__global__ __launch_bounds__(128)
void sp_gemm_logA(const float* __restrict__ x,
                  const unsigned short* __restrict__ anbf_u,
                  float* __restrict__ logA) {
  const int b    = blockIdx.y;
  const int tid  = threadIdx.x;
  const int wave = tid >> 5;
  const int lane = tid & 31;
  const int col  = lane & 15;       // N-index inside a 16x16 tile
  const int hi   = lane >> 4;       // half-wave select
  const int rowBase = blockIdx.x * 64 + wave * 16;

  __shared__ float lds_s[4][16][65];   // +1 pad: stride 65 -> conflict-free
  __shared__ float lds_norm[4][16];

  if (rowBase < NPOS) {
    const __bf16* anbf = reinterpret_cast<const __bf16*>(anbf_u);
    const int   arow = rowBase + col;                 // A-fragment row = lane%16
    const float* xrow = x + ((size_t)b * NPOS + arow) * CCH;
    const int r1   = hi << 3;   // unused: 0 (lanes 0-15) / 8 (lanes 16-31)
    const int koff = hi << 4;   // B K-half:    0 / 16

    v8f acc[4] = {};
    float ss = 0.f;                                   // per-lane sum-of-squares

    #pragma unroll 4
    for (int kb = 0; kb < 16; ++kb) {                 // C in steps of 32
      const int cb = kb << 5;
      __builtin_prefetch(xrow + cb + 512, 0, 1);      // global_prefetch_b8
      // A fragment: 16 consecutive K-values (matches B): K = 16*hi + s.
      v16f vf;
      float4* pv = reinterpret_cast<float4*>(&vf);
      pv[0] = *(const float4*)(xrow + cb + koff);
      pv[1] = *(const float4*)(xrow + cb + koff + 4);
      pv[2] = *(const float4*)(xrow + cb + koff + 8);
      pv[3] = *(const float4*)(xrow + cb + koff + 12);
      #pragma unroll
      for (int j = 0; j < 16; ++j) ss = fmaf(vf[j], vf[j], ss);
      // RNE float->bf16; backend picks native packed cvt when available.
      const v16bf a = __builtin_convertvector(vf, v16bf);
      #pragma unroll
      for (int t = 0; t < 4; ++t) {   // 4 N-tiles of 16 anchors
        // B fragment: 16 consecutive K-values (channels) of one anchor column.
        const v16bf bf =
            *(const v16bf*)(anbf + ((16 * t + col) * CCH + cb + koff));
        acc[t] = __builtin_amdgcn_wmma_f32_16x16x32_bf16(
            false, a, false, bf, (short)0, acc[t], false, false);
      }
    }

    // Full row norm: lane l and l+16 loaded complementary halves of row l%16.
    const float ss_tot = ss + __shfl_xor(ss, 16, 32);
    if (hi == 0) lds_norm[wave][col] = sqrtf(ss_tot);

    // C/D layout: lanes 0-15 VGPR r -> M=r ; lanes 16-31 VGPR r -> M=8+r.
    #pragma unroll
    for (int t = 0; t < 4; ++t)
      #pragma unroll
      for (int r = 0; r < 8; ++r)
        lds_s[wave][r + (hi << 3)][(t << 4) + col] = acc[t][r];
  }
  __syncthreads();

  // Per-row logsumexp over 64 anchors: logA = LSE_k(20*cos_k - 20).
  if (tid < 64) {
    const int row_g = blockIdx.x * 64 + tid;
    if (row_g < NPOS) {
      const int wv = tid >> 4, rl = tid & 15;
      const float inv = 1.f / (lds_norm[wv][rl] + 1e-12f);
      float mx = -3.402823e38f;
      for (int k = 0; k < NANCH; ++k)
        mx = fmaxf(mx, 20.f * (lds_s[wv][rl][k] * inv) - 20.f);
      float sum = 0.f;
      for (int k = 0; k < NANCH; ++k)
        sum += expf(20.f * (lds_s[wv][rl][k] * inv) - 20.f - mx);
      logA[b * NPOS + row_g] = mx + logf(sum);
    }
  }
}

// ------------------- 3) scalar Sinkhorn recursion per batch ----------------
__global__ void sp_sinkhorn(const float* __restrict__ logA,
                            float* __restrict__ wbuf) {
  const int b   = blockIdx.x;
  const int tid = threadIdx.x;
  float la[4];
  #pragma unroll
  for (int j = 0; j < 4; ++j) {
    const int idx = tid + (j << 8);
    la[j] = (idx < NPOS) ? logA[b * NPOS + idx] : -1e30f;  // pad -> sigmoid 0
  }
  __shared__ float red[8];
  float v = 0.f;
  const float log_mu = logf(0.5f);
  for (int it = 0; it < 10; ++it) {
    float ls = 0.f;
    #pragma unroll
    for (int j = 0; j < 4; ++j) {
      const float z = v * EPS_INV + la[j];
      ls += 1.f / (1.f + expf(-z));     // transported column mass
    }
    #pragma unroll
    for (int off = 16; off > 0; off >>= 1) ls += __shfl_xor(ls, off, 32);
    if ((tid & 31) == 0) red[tid >> 5] = ls;
    __syncthreads();
    float tot = 0.f;
    #pragma unroll
    for (int w = 0; w < 8; ++w) tot += red[w];   // replicated, deterministic
    __syncthreads();
    v += 0.1f * (log_mu - logf(tot * (1.f / 784.f)));
  }
  // mixing/mu folded with 1/N of the final mean: w = sigmoid(z)/(mu*N)
  #pragma unroll
  for (int j = 0; j < 4; ++j) {
    const int idx = tid + (j << 8);
    if (idx < NPOS) {
      const float z = v * EPS_INV + la[j];
      wbuf[b * NPOS + idx] = (1.f / (1.f + expf(-z))) * (1.f / (0.5f * 784.f));
    }
  }
}

// ------------------- 4) chunked weighted sum over pixels -------------------
__global__ void sp_wsum_partial(const float* __restrict__ x,
                                const float* __restrict__ wbuf,
                                float* __restrict__ part) {
  const int b     = blockIdx.y;
  const int chunk = blockIdx.x;        // 4 chunks of 196 rows
  const int c0    = threadIdx.x;       // channels c0 and c0+256
  float a0 = 0.f, a1 = 0.f;
  const int rbeg = chunk * 196, rend = rbeg + 196;
  for (int r = rbeg; r < rend; ++r) {
    const float wv = wbuf[b * NPOS + r];
    const float* xp = x + ((size_t)(b * NPOS + r) << 9);
    a0 = fmaf(wv, xp[c0], a0);
    a1 = fmaf(wv, xp[c0 + 256], a1);
  }
  float* pp = part + (((size_t)b * 4 + chunk) << 9);
  pp[c0]       = a0;
  pp[c0 + 256] = a1;
}

// ------------------- 5) deterministic chunk reduce -------------------------
__global__ void sp_reduce_out(const float* __restrict__ part,
                              float* __restrict__ out) {
  const int idx = blockIdx.x * 256 + threadIdx.x;   // 0..65535 = b*512 + c
  const int b = idx >> 9;
  const int c = idx & 511;
  const float* pp = part + (((size_t)b * 4) << 9) + c;
  out[idx] = pp[0] + pp[512] + pp[1024] + pp[1536];
}

// ---------------------------------------------------------------------------
extern "C" void kernel_launch(void* const* d_in, const int* in_sizes, int n_in,
                              void* d_out, int out_size, void* d_ws, size_t ws_size,
                              hipStream_t stream) {
  (void)in_sizes; (void)n_in; (void)out_size; (void)ws_size;
  const float* x    = (const float*)d_in[0];   // bow_feats (128,28,28,512) f32
  const float* kern = (const float*)d_in[1];   // kernel (64,512) f32
  float* out = (float*)d_out;                  // emb (128,512) f32

  char* ws = (char*)d_ws;                      // ~1.83 MB total
  unsigned short* anbf = (unsigned short*)ws;               //  65536 B bf16 anchors
  float* logA = (float*)(ws + 65536);                       // 401408 B
  float* wbuf = (float*)(ws + 65536 + 401408);              // 401408 B
  float* part = (float*)(ws + 65536 + 2 * 401408);          // 1 MB partials

  sp_norm_anchors<<<64, 256, 0, stream>>>(kern, anbf);
  sp_gemm_logA<<<dim3(13, 128), 128, 0, stream>>>(x, anbf, logA);
  sp_sinkhorn<<<128, 256, 0, stream>>>(logA, wbuf);
  sp_wsum_partial<<<dim3(4, 128), 256, 0, stream>>>(x, wbuf, part);
  sp_reduce_out<<<256, 256, 0, stream>>>(part, out);
}
